// MetaController_40647570489700
// MI455X (gfx1250) — matrix operand order repaired and checked by reading, not compiled, not run on hardware
//
#include <hip/hip_runtime.h>
#include <hip/hip_bf16.h>
#include <math.h>

// Problem constants
#define DD   1024
#define HH   2048
#define RR   16
#define MM   4096            // B*S tokens
#define OUTC 32768           // 2*R*D
#define WAC  16384           // D*R  (wa columns / wb offset)

typedef __attribute__((ext_vector_type(16))) __bf16 v16bf;
typedef __attribute__((ext_vector_type(8)))  __bf16 v8bf;
typedef __attribute__((ext_vector_type(8)))  float  v8f;
typedef __attribute__((ext_vector_type(4)))  unsigned v4u;
typedef __attribute__((ext_vector_type(8)))  int      v8i;
typedef __attribute__((ext_vector_type(4)))  int      v4i;

// LDS staging geometry for wa_fused A-chunks: 32 rows x 128 k bf16, padded.
// Row stride 136 elements (272B) -> half-wave lanes land on distinct banks.
#define KCHUNK   128
#define AROWS    32
#define ASTRIDE  136                       // elements
#define ABUFELEM (AROWS * ASTRIDE)         // 4352 elements = 8704 bytes

__device__ __forceinline__ __bf16 f2bf(float f) {
  unsigned u = __builtin_bit_cast(unsigned, f);
  u += 0x7fffu + ((u >> 16) & 1u);                 // round-to-nearest-even
  return __builtin_bit_cast(__bf16, (unsigned short)(u >> 16));
}

// A fragment (16x32 bf16): lane m = l&15; half l>>4 selects K 0-7/16-23 vs 8-15/24-31
__device__ __forceinline__ v16bf loadA(const __bf16* p, int ld, int row, int k0, int lane) {
  int half = (lane >> 4) & 1;
  const __bf16* b = p + (size_t)row * ld + k0 + half * 8;
  v8bf lo = *(const v8bf*)b;
  v8bf hi = *(const v8bf*)(b + 16);
  return __builtin_shufflevector(lo, hi, 0,1,2,3,4,5,6,7,8,9,10,11,12,13,14,15);
}

// Same A fragment sourced from a padded LDS chunk (row-local, k-local)
__device__ __forceinline__ v16bf loadA_lds(const __bf16* p, int row, int kl, int lane) {
  int half = (lane >> 4) & 1;
  const __bf16* b = p + row * ASTRIDE + kl + half * 8;
  v8bf lo = *(const v8bf*)b;
  v8bf hi = *(const v8bf*)(b + 16);
  return __builtin_shufflevector(lo, hi, 0,1,2,3,4,5,6,7,8,9,10,11,12,13,14,15);
}

// B fragment (32x16 bf16): lane n = l&15; half l>>4 selects K 0-7/16-23 vs 8-15/24-31
__device__ __forceinline__ v16bf loadB(const __bf16* p, int ld, int col, int k0, int lane) {
  int half = (lane >> 4) & 1;
  const __bf16* b = p + (size_t)col * ld + k0 + half * 8;
  v8bf lo = *(const v8bf*)b;
  v8bf hi = *(const v8bf*)(b + 16);
  return __builtin_shufflevector(lo, hi, 0,1,2,3,4,5,6,7,8,9,10,11,12,13,14,15);
}

// ---------------- TDM: 2D tile load (rows x KCHUNK bf16) into padded LDS ----------------
#if defined(__has_builtin)
#if __has_builtin(__builtin_amdgcn_tensor_load_to_lds) && __has_builtin(__builtin_amdgcn_s_wait_tensorcnt)
#define USE_TDM 1
#endif
#endif

#ifdef USE_TDM
__device__ __forceinline__ void tdm_load_a(unsigned lds_off, const __bf16* gptr) {
  unsigned long long ga = (unsigned long long)(size_t)gptr;
  v4u g0;
  g0[0] = 1u;                                   // count=1, user descriptor
  g0[1] = lds_off;                              // lds_addr
  g0[2] = (unsigned)ga;                         // global_addr[31:0]
  g0[3] = ((unsigned)(ga >> 32) & 0x1FFFFFFu)   // global_addr[56:32]
          | (2u << 30);                         // type=2 ("image")
  v8i g1;
  // data_size=1 (2B) | pad_enable | pad_interval=5 (64 DW = one 128-elem row)
  // | pad_amount=3 (4 DW = 16B pad -> 272B row stride)
  g1[0] = (int)((1u << 16) | (1u << 20) | (5u << 22) | (3u << 25));
  g1[1] = (int)((HH & 0xFFFFu) << 16);          // tensor_dim0[15:0] = 2048
  g1[2] = (int)(0x8000u << 16);                 // tensor_dim1[15:0] = 32768 (no OOB)
  g1[3] = (int)((unsigned)KCHUNK << 16);        // tile_dim0 = 128
  g1[4] = (int)AROWS;                           // tile_dim1 = 32, tile_dim2 = 0
  g1[5] = (int)HH;                              // tensor_dim0_stride = 2048
  g1[6] = 0;
  g1[7] = 0;
  v4i gz4 = {0, 0, 0, 0};                       // 2D tensor: groups 2/3 unused
  v8i gz8 = {0, 0, 0, 0, 0, 0, 0, 0};
  __builtin_amdgcn_tensor_load_to_lds(g0, g1, gz4, gz4, gz8, 0);
}
#endif

// ---------------- prep kernels ----------------

__global__ __launch_bounds__(256) void cvt_bf16(const float* in, __bf16* out, int n) {
  int i = blockIdx.x * 256 + threadIdx.x;
  if (i < n) out[i] = f2bf(in[i]);
}

// (rows x cols) fp32, row-stride ldin  ->  (cols x rows) bf16, row-stride ldout
__global__ __launch_bounds__(256) void transpose_cvt(const float* in, long ldin,
                                                     __bf16* out, long ldout) {
  __shared__ float tile[32][33];
  int tx = threadIdx.x, ty = threadIdx.y;
  long x  = (long)blockIdx.x * 32 + tx;
  long y0 = (long)blockIdx.y * 32;
#pragma unroll
  for (int j = 0; j < 4; ++j) tile[ty + 8 * j][tx] = in[(y0 + ty + 8 * j) * ldin + x];
  __syncthreads();
  long ox  = (long)blockIdx.y * 32 + tx;
  long oy0 = (long)blockIdx.x * 32;
#pragma unroll
  for (int j = 0; j < 4; ++j) out[(oy0 + ty + 8 * j) * ldout + ox] = f2bf(tile[tx][ty + 8 * j]);
}

// W1bsumT[r][k] = sum_d w1[k, WAC + d*16 + r]   (stored transposed, bf16, ld=HH)
__global__ __launch_bounds__(256) void wb_colsum(const float* w1, __bf16* WbT) {
  __shared__ float red[256];
  int k = blockIdx.x, tid = threadIdx.x;
  int r = tid & 15, part = tid >> 4;
  const float* src = w1 + (size_t)k * OUTC + WAC;
  float acc = 0.f;
  for (int d = part; d < DD; d += 16) acc += src[d * RR + r];
  red[tid] = acc;
  __syncthreads();
  if (tid < 16) {
    float s = 0.f;
#pragma unroll
    for (int j = 0; j < 16; ++j) s += red[tid + 16 * j];
    WbT[(size_t)tid * HH + k] = f2bf(s);
  }
}

__global__ void b1_sum(const float* b1, float* B1s) {
  int r = threadIdx.x;
  if (r < 16) {
    float a = 0.f;
    for (int d = 0; d < DD; ++d) a += b1[WAC + d * RR + r];
    B1s[r] = a;
  }
}

// ---------------- GEMM1: h = silu(latent @ w0 + b0), bf16 out ----------------
// Wave tile 32Mx64N (2x4 WMMA tiles): 8 wmma per 6 fragment loads.
__global__ __launch_bounds__(256) void gemm1_silu(const __bf16* A, const __bf16* Bt,
                                                  const float* b0, __bf16* H) {
  int lane = threadIdx.x & 31, wave = threadIdx.x >> 5;
  int idx = lane & 15, half = lane >> 4;
  int m0 = blockIdx.y * 64 + (wave >> 2) * 32;
  int n0 = blockIdx.x * 256 + (wave & 3) * 64;
  v8f c[2][4] = {};
  int row0 = m0 + idx, row1 = m0 + 16 + idx;
  for (int k0 = 0; k0 < DD; k0 += 32) {
    v16bf a0 = loadA(A, DD, row0, k0, lane);
    v16bf a1 = loadA(A, DD, row1, k0, lane);
#pragma unroll
    for (int t = 0; t < 4; ++t) {
      v16bf b = loadB(Bt, DD, n0 + 16 * t + idx, k0, lane);
      c[0][t] = __builtin_amdgcn_wmma_f32_16x16x32_bf16(false, a0, false, b, (short)0, c[0][t], false, false);
      c[1][t] = __builtin_amdgcn_wmma_f32_16x16x32_bf16(false, a1, false, b, (short)0, c[1][t], false, false);
    }
  }
#pragma unroll
  for (int t = 0; t < 4; ++t) {
    int n = n0 + 16 * t + idx;
    float bias = b0[n];
#pragma unroll
    for (int mi = 0; mi < 2; ++mi) {
#pragma unroll
      for (int i = 0; i < 8; ++i) {
        int token = m0 + mi * 16 + half * 8 + i;
        float x = c[mi][t][i] + bias;
        float h = x / (1.0f + __expf(-x));      // silu
        H[(size_t)token * HH + n] = f2bf(h);
      }
    }
  }
}

// ---------------- s = h @ W1bsumT + b1sum  (4096 x 16, fp32) ----------------
__global__ __launch_bounds__(256) void s_gemm(const __bf16* Hb, const __bf16* WbT,
                                              const float* B1s, float* S) {
  int lane = threadIdx.x & 31, wave = threadIdx.x >> 5;
  int idx = lane & 15, half = lane >> 4;
  int m0 = (blockIdx.x * 8 + wave) * 16;
  v8f c = {};
  int row = m0 + idx;
  for (int k0 = 0; k0 < HH; k0 += 32) {
    v16bf a = loadA(Hb, HH, row, k0, lane);
    v16bf b = loadB(WbT, HH, idx, k0, lane);
    c = __builtin_amdgcn_wmma_f32_16x16x32_bf16(false, a, false, b, (short)0, c, false, false);
  }
  float bias = B1s[idx];
#pragma unroll
  for (int i = 0; i < 8; ++i)
    S[(size_t)(m0 + half * 8 + i) * RR + idx] = c[i] + bias;
}

// ------- fused wa-GEMM + (wa·s) contraction + residual scale, fp32 out -------
// Block: 32 tokens x 32 d.  Wave: 32 tokens x 4 d  (2x4 WMMA tiles).
// A chunks (32 x 128 bf16) staged to LDS by the Tensor Data Mover, double-buffered
// and pipelined on TENSORcnt; B streams from global (pre-transposed, contiguous-K).
__global__ __launch_bounds__(256) void wa_fused(const __bf16* Hb, const __bf16* WaT,
                                                const float* b1, const float* S,
                                                const float* latent, float* out) {
  __shared__ __bf16 abuf[2][ABUFELEM];
  int lane = threadIdx.x & 31, wave = threadIdx.x >> 5;
  int idx = lane & 15, half = lane >> 4;
  int m0 = blockIdx.y * 32;
  int d0 = blockIdx.x * 32 + wave * 4;
  v8f c[2][4] = {};

  const int nchunk = HH / KCHUNK;               // 16
#ifdef USE_TDM
  if (wave == 0)
    tdm_load_a((unsigned)(size_t)&abuf[0][0], Hb + (size_t)m0 * HH);
#endif
  for (int cck = 0; cck < nchunk; ++cck) {
    const __bf16* acur = &abuf[cck & 1][0];
#ifdef USE_TDM
    if (wave == 0) __builtin_amdgcn_s_wait_tensorcnt(0);   // chunk cck landed
    __syncthreads();                                       // visible to all; prev buf free
    if (wave == 0 && cck + 1 < nchunk)
      tdm_load_a((unsigned)(size_t)&abuf[(cck + 1) & 1][0],
                 Hb + (size_t)m0 * HH + (size_t)(cck + 1) * KCHUNK);
#else
    __syncthreads();
    {   // cooperative copy fallback: 256 threads, 32 rows x 128 elems (16 bf16 each)
      int tid = threadIdx.x;
      int r = tid >> 3, kk = (tid & 7) * 16;
      const v8bf* src = (const v8bf*)(Hb + (size_t)(m0 + r) * HH + cck * KCHUNK + kk);
      *(v8bf*)(&abuf[cck & 1][r * ASTRIDE + kk]) = src[0];
      *(v8bf*)(&abuf[cck & 1][r * ASTRIDE + kk + 8]) = src[1];
    }
    __syncthreads();
#endif
#pragma unroll
    for (int ks = 0; ks < KCHUNK / 32; ++ks) {
      int kg = cck * KCHUNK + ks * 32;
      v16bf a0 = loadA_lds(acur, idx,      ks * 32, lane);
      v16bf a1 = loadA_lds(acur, 16 + idx, ks * 32, lane);
#pragma unroll
      for (int t = 0; t < 4; ++t) {
        v16bf b = loadB(WaT, HH, (d0 + t) * RR + idx, kg, lane);
        c[0][t] = __builtin_amdgcn_wmma_f32_16x16x32_bf16(false, a0, false, b, (short)0, c[0][t], false, false);
        c[1][t] = __builtin_amdgcn_wmma_f32_16x16x32_bf16(false, a1, false, b, (short)0, c[1][t], false, false);
      }
    }
#ifdef USE_TDM
    __syncthreads();                                       // done reading buf before TDM overwrites
#endif
  }

  float sv[2][8];
#pragma unroll
  for (int mi = 0; mi < 2; ++mi)
#pragma unroll
    for (int i = 0; i < 8; ++i)
      sv[mi][i] = S[(size_t)(m0 + mi * 16 + half * 8 + i) * RR + idx];
#pragma unroll
  for (int t = 0; t < 4; ++t) {
    float bw = b1[(d0 + t) * RR + idx];
#pragma unroll
    for (int mi = 0; mi < 2; ++mi) {
#pragma unroll
      for (int i = 0; i < 8; ++i) {
        float p = (c[mi][t][i] + bw) * sv[mi][i];  // wa[token,d,r]*s[token,r], r = idx
        p += __shfl_xor(p, 8, 16);                 // reduce over 16 r-lanes (per half-wave)
        p += __shfl_xor(p, 4, 16);
        p += __shfl_xor(p, 2, 16);
        p += __shfl_xor(p, 1, 16);
        if (idx == t) {
          int token = m0 + mi * 16 + half * 8 + i;
          size_t o = (size_t)token * DD + (d0 + t);
          out[o] = latent[o] * (1.0f + p);         // latent + latent*control
        }
      }
    }
  }
}

// ---------------- launcher ----------------
extern "C" void kernel_launch(void* const* d_in, const int* in_sizes, int n_in,
                              void* d_out, int out_size, void* d_ws, size_t ws_size,
                              hipStream_t stream) {
  const float* latent = (const float*)d_in[0];
  const float* w0     = (const float*)d_in[1];
  const float* b0     = (const float*)d_in[2];
  const float* w1     = (const float*)d_in[3];
  const float* b1     = (const float*)d_in[4];
  float* out = (float*)d_out;
  char* ws = (char*)d_ws;

  // workspace layout (bytes)
  __bf16* latB  = (__bf16*)(ws + 0);                        // 4096x1024   (8 MiB)
  __bf16* w0T   = (__bf16*)(ws + 8388608);                  // 2048x1024   (4 MiB)
  __bf16* w1aT  = (__bf16*)(ws + 12582912);                 // 16384x2048  (64 MiB)
  __bf16* wbT   = (__bf16*)(ws + 79691776);                 // 16x2048
  float*  b1s   = (float*) (ws + 79757568);                 // 16
  __bf16* hB    = (__bf16*)(ws + 79758080);                 // 4096x2048   (16 MiB)
  float*  sF    = (float*) (ws + 96535296);                 // 4096x16

  // --- prep ---
  cvt_bf16<<<(MM * DD) / 256, 256, 0, stream>>>(latent, latB, MM * DD);
  {
    dim3 blk(32, 8);
    transpose_cvt<<<dim3(HH / 32, DD / 32), blk, 0, stream>>>(w0, HH, w0T, DD);       // w0^T
    transpose_cvt<<<dim3(WAC / 32, HH / 32), blk, 0, stream>>>(w1, OUTC, w1aT, HH);   // w1a^T
  }
  wb_colsum<<<HH, 256, 0, stream>>>(w1, wbT);
  b1_sum<<<1, 16, 0, stream>>>(b1, b1s);

  // --- GEMM1 + silu ---
  gemm1_silu<<<dim3(HH / 256, MM / 64), 256, 0, stream>>>(latB, w0T, b0, hB);

  // --- s = h @ W1bsum ---
  s_gemm<<<MM / 128, 256, 0, stream>>>(hB, wbT, b1s, sF);

  // --- fused wa GEMM + contraction + residual (TDM-staged A operand) ---
  wa_fused<<<dim3(DD / 32, MM / 32), 256, 0, stream>>>(hB, w1aT, b1, sF, latent, out);
}
